// Model_60447369724580
// MI455X (gfx1250) — matrix operand order repaired and checked by reading, not compiled, not run on hardware
//
#include <hip/hip_runtime.h>
#include <hip/hip_bf16.h>
#include <math.h>

typedef __attribute__((ext_vector_type(16))) _Float16 v16h;
typedef __attribute__((ext_vector_type(8)))  _Float16 v8h;
typedef __attribute__((ext_vector_type(8)))  float    v8f;

#define RROWS 2568   // BATCH * ENC_IN = 8 * 321
#define SEQL  512
#define PREDL 96
#define ENC   321
#define NBATCH 8
#define NDBC  288    // padded dt_rank + 2*D_STATE projection width (multiple of 32)

__device__ __forceinline__ float silu_f(float v)     { return v / (1.0f + __expf(-v)); }
__device__ __forceinline__ float softplus_f(float v) { return (v > 20.0f) ? v : log1pf(__expf(v)); }

// ---------------------------------------------------------------------------
// fp32 -> fp16 weight conversion with optional zero padding (dst_n >= src_n)
// ---------------------------------------------------------------------------
__global__ void k_cvt_f16(_Float16* __restrict__ dst, const float* __restrict__ src,
                          int n_dst, int n_src) {
  int i = blockIdx.x * blockDim.x + threadIdx.x;
  if (i < n_dst) dst[i] = (i < n_src) ? (_Float16)src[i] : (_Float16)0.0f;
}

// ---------------------------------------------------------------------------
// RevIN: per (b,c) row -> mean/std over L=512, normalized row written as f16
// one wave32 per row; each lane owns 16 strided elements
// ---------------------------------------------------------------------------
__global__ void k_revin(const float* __restrict__ x, const float* __restrict__ rw,
                        const float* __restrict__ rb, _Float16* __restrict__ xn16,
                        float* __restrict__ meanv, float* __restrict__ stdv) {
  int row = blockIdx.x;                 // row = b*ENC + c
  int b = row / ENC, c = row % ENC;
  int lane = threadIdx.x;
  const float* px = x + (size_t)b * SEQL * ENC + c;
  float vals[16];
  float s = 0.f, s2 = 0.f;
  for (int i = 0; i < 16; ++i) {
    float v = px[(size_t)(lane + 32 * i) * ENC];
    vals[i] = v; s += v; s2 += v * v;
  }
  for (int off = 16; off; off >>= 1) {
    s  += __shfl_xor(s,  off, 32);
    s2 += __shfl_xor(s2, off, 32);
  }
  float mean = s * (1.0f / SEQL);
  float var  = s2 * (1.0f / SEQL) - mean * mean;
  float sd   = sqrtf(var + 1e-5f);
  if (lane == 0) { meanv[row] = mean; stdv[row] = sd; }
  float w = rw[c], bb = rb[c], inv = 1.0f / sd;
  for (int i = 0; i < 16; ++i)
    xn16[(size_t)row * SEQL + lane + 32 * i] = (_Float16)((vals[i] - mean) * inv * w + bb);
}

// ---------------------------------------------------------------------------
// Register-blocked WMMA GEMM: C(MxN,f32) = A(MxK,f16) @ W(NxK,f16)^T + bias
// One wave32 computes a (16*MT) x (16*WN) macro-tile -> MT*WN accumulators.
// A fragments reused WN times, B fragments reused MT times.
// Fragment element maps per CDNA5 16-bit WMMA layouts (h = lane>>4):
//   A lane: row m = lane&15, K chunks [k0+8h,+8) and [k0+16+8h,+8)
//   B lane: row n = lane&15 of W (K-major), K chunks [k0+8h,+8) and [k0+16+8h,+8)
//   D lane: col n = lane&15, rows r+8h, r=0..7
// ---------------------------------------------------------------------------
template <int MT, int WN>
__global__ __launch_bounds__(32) void k_gemm_wmma(
    const _Float16* __restrict__ A, const _Float16* __restrict__ W,
    const float* __restrict__ bias, float* __restrict__ Cf, _Float16* __restrict__ Ch,
    int M, int N, int K) {
  int Ng = N / (16 * WN);
  int tile = blockIdx.x;
  int mg = tile / Ng, ng = tile % Ng;
  int lane = threadIdx.x;
  int half = lane >> 4;
  int l16  = lane & 15;
  const _Float16* arow[MT];
  const _Float16* brow[WN];
#pragma unroll
  for (int i = 0; i < MT; ++i) {
    int m = mg * 16 * MT + i * 16 + l16;
    if (m >= M) m = M - 1;              // clamp (stores are guarded)
    arow[i] = A + (size_t)m * K;
  }
#pragma unroll
  for (int j = 0; j < WN; ++j) {
    int n = ng * 16 * WN + j * 16 + l16;
    brow[j] = W + (size_t)n * K;
  }
  v8f acc[MT][WN];
  v8f zero = {};
#pragma unroll
  for (int i = 0; i < MT; ++i)
#pragma unroll
    for (int j = 0; j < WN; ++j) acc[i][j] = zero;

  for (int k0 = 0; k0 < K; k0 += 32) {
    v16h av[MT], bv[WN];
#pragma unroll
    for (int i = 0; i < MT; ++i) {
      v8h a0 = *(const v8h*)(arow[i] + k0 + 8 * half);
      v8h a1 = *(const v8h*)(arow[i] + k0 + 16 + 8 * half);
      av[i] = __builtin_shufflevector(a0, a1, 0,1,2,3,4,5,6,7,8,9,10,11,12,13,14,15);
    }
#pragma unroll
    for (int j = 0; j < WN; ++j) {
      v8h b0 = *(const v8h*)(brow[j] + k0 + 8 * half);
      v8h b1 = *(const v8h*)(brow[j] + k0 + 16 + 8 * half);
      bv[j] = __builtin_shufflevector(b0, b1, 0,1,2,3,4,5,6,7,8,9,10,11,12,13,14,15);
    }
#pragma unroll
    for (int i = 0; i < MT; ++i)
#pragma unroll
      for (int j = 0; j < WN; ++j)
        acc[i][j] = __builtin_amdgcn_wmma_f32_16x16x32_f16(false, av[i], false, bv[j],
                                                           (short)0, acc[i][j], false, false);
  }
#pragma unroll
  for (int j = 0; j < WN; ++j) {
    int n = ng * 16 * WN + j * 16 + l16;
    float bn = bias ? bias[n] : 0.0f;
#pragma unroll
    for (int i = 0; i < MT; ++i) {
#pragma unroll
      for (int r = 0; r < 8; ++r) {
        int mm = mg * 16 * MT + i * 16 + r + 8 * half;
        if (mm < M) {
          float v = acc[i][j][r] + bn;
          if (Cf) Cf[(size_t)mm * N + n] = v;
          if (Ch) Ch[(size_t)mm * N + n] = (_Float16)v;
        }
      }
    }
  }
}

// ---------------------------------------------------------------------------
// L==1 mamba, stage 1: xa = silu(xc*conv_w[:,1] + conv_b)   (K=2 conv, xpad[0]=0)
// xz rows are [xc(0..D-1) | z(D..2D-1)]
// ---------------------------------------------------------------------------
__global__ void k_conv_silu(const float* __restrict__ xz, const float* __restrict__ conv_w,
                            const float* __restrict__ conv_b, float* __restrict__ xa,
                            _Float16* __restrict__ xa16, int R, int D) {
  int i = blockIdx.x * blockDim.x + threadIdx.x;
  if (i >= R * D) return;
  int r = i / D, d = i - r * D;
  float v = xz[(size_t)r * 2 * D + d] * conv_w[d * 2 + 1] + conv_b[d];
  float sv = silu_f(v);
  xa[i] = sv; xa16[i] = (_Float16)sv;
}

// ---------------------------------------------------------------------------
// L==1 mamba, stage 2 (collapsed scan):
//   dt_d = softplus(dbc[:,0:rank] @ dt_w.T + dt_b);  bc = B . C
//   y_d  = (dt_d*xa_d*bc + xa_d*D_d) * silu(z_d)
// one wave32 per row; dbc row = [dt(rank) | B(128) | C(128) | pad]
// ---------------------------------------------------------------------------
__global__ void k_finalize_l1(const float* __restrict__ dbc, const float* __restrict__ xa,
                              const float* __restrict__ xz, const float* __restrict__ dt_w,
                              const float* __restrict__ dt_b, const float* __restrict__ Dp,
                              _Float16* __restrict__ y16, int R, int D, int rank) {
  int row = blockIdx.x; if (row >= R) return;
  int lane = threadIdx.x;
  const float* db = dbc + (size_t)row * NDBC;
  float bc = 0.f;
  for (int s = lane; s < 128; s += 32) bc += db[rank + s] * db[rank + 128 + s];
  for (int off = 16; off; off >>= 1) bc += __shfl_xor(bc, off, 32);
  float dtv[16];
  for (int j = 0; j < rank; ++j) dtv[j] = db[j];
  int per = D / 32;
  for (int q = 0; q < per; ++q) {
    int d = lane + 32 * q;
    float acc = dt_b[d];
    for (int j = 0; j < rank; ++j) acc += dtv[j] * dt_w[d * rank + j];
    float dt  = softplus_f(acc);
    float xad = xa[(size_t)row * D + d];
    float y   = dt * xad * bc + xad * Dp[d];
    float z   = xz[(size_t)row * 2 * D + D + d];
    y *= silu_f(z);
    y16[(size_t)row * D + d] = (_Float16)y;
  }
}

// ---------------------------------------------------------------------------
// d_model==1 mamba (m1/m4): true sequential selective scan over L steps.
// One wave32 per row, 4 of the 128 states per lane, shuffle-reduce y each step.
// Optionally adds `addend` (x3) and writes f16 into a strided buffer (concat).
// ---------------------------------------------------------------------------
__global__ void k_mamba_seq(const float* __restrict__ U, int L,
                            const float* __restrict__ in_w,  const float* __restrict__ conv_w,
                            const float* __restrict__ conv_b,const float* __restrict__ xproj_w,
                            const float* __restrict__ dt_w,  const float* __restrict__ dt_b,
                            const float* __restrict__ A_log, const float* __restrict__ Dp,
                            const float* __restrict__ out_w, const float* __restrict__ addend,
                            float* __restrict__ outf, _Float16* __restrict__ outh,
                            int ldh, int coff, int R) {
  int row = blockIdx.x; if (row >= R) return;
  int lane = threadIdx.x;
  float As[4], Bs[4], Cs[4], h[4];
  for (int q = 0; q < 4; ++q) {
    int s = lane * 4 + q;
    As[q] = -__expf(A_log[s]);
    Bs[q] = xproj_w[1 + s];
    Cs[q] = xproj_w[1 + 128 + s];
    h[q]  = 0.f;
  }
  float wi0 = in_w[0], wi1 = in_w[1];
  float c0 = conv_w[0], c1 = conv_w[1], cb = conv_b[0];
  float xp0 = xproj_w[0], dw = dt_w[0], dbias = dt_b[0], Dv = Dp[0], ow = out_w[0];
  const float* u = U + (size_t)row * L;
  float prev_xc = 0.f;
  for (int l = 0; l < L; ++l) {
    float ul = u[l];
    float xc = ul * wi0;
    float zl = ul * wi1;
    float xcv = prev_xc * c0 + xc * c1 + cb;
    prev_xc = xc;
    float xa = silu_f(xcv);
    float dt = softplus_f(xa * xp0 * dw + dbias);
    float dxx = dt * xa * xa;                     // dt * x_t, with B_t = xa * xpB
    float part = 0.f;
    for (int q = 0; q < 4; ++q) {
      h[q] = h[q] * __expf(dt * As[q]) + dxx * Bs[q];
      part += h[q] * Cs[q];
    }
    for (int off = 16; off; off >>= 1) part += __shfl_xor(part, off, 32);
    if (lane == 0) {
      float y = (xa * part + xa * Dv) * silu_f(zl) * ow;
      if (addend) y += addend[(size_t)row * L + l];
      if (outf) outf[(size_t)row * L + l] = y;
      if (outh) outh[(size_t)row * ldh + coff + l] = (_Float16)y;
    }
  }
}

// ---------------------------------------------------------------------------
// small fused elementwise kernels
// ---------------------------------------------------------------------------
__global__ void k_sum3_f16(const float* __restrict__ a, const float* __restrict__ b,
                           const float* __restrict__ c, _Float16* __restrict__ o, int n) {
  int i = blockIdx.x * blockDim.x + threadIdx.x;
  if (i < n) o[i] = (_Float16)(a[i] + b[i] + c[i]);
}

// concat[:, 0:256] = f16(t + h1)   (concat row stride 512)
__global__ void k_addres(const float* __restrict__ t, const float* __restrict__ h1,
                         _Float16* __restrict__ cat, int R) {
  int i = blockIdx.x * blockDim.x + threadIdx.x;
  if (i >= R * 256) return;
  int r = i >> 8, c = i & 255;
  cat[(size_t)r * 512 + c] = (_Float16)(t[i] + h1[i]);
}

// de-RevIN + (BC,96) -> (B,96,C) transpose
__global__ void k_final(const float* __restrict__ ol, const float* __restrict__ rw,
                        const float* __restrict__ rb, const float* __restrict__ meanv,
                        const float* __restrict__ stdv, float* __restrict__ out) {
  int i = blockIdx.x * blockDim.x + threadIdx.x;
  if (i >= NBATCH * PREDL * ENC) return;
  int c = i % ENC;
  int t = (i / ENC) % PREDL;
  int b = i / (ENC * PREDL);
  int row = b * ENC + c;
  float v = ol[(size_t)row * PREDL + t];
  out[i] = (v - rb[c]) / (rw[c] + 1e-10f) * stdv[row] + meanv[row];
}

// ===========================================================================
// host side
// ===========================================================================
struct MP { const float *in_w,*conv_w,*conv_b,*xproj_w,*dt_w,*dt_b,*A_log,*D,*out_w; };
struct Ptrs {
  const float *x,*revin_w,*revin_b;
  const float *lin1_w,*lin1_b,*lin2_w,*lin2_b,*lin3_w,*lin3_b,*lin4_w,*lin4_b;
  MP m1,m2,m3,m4;
};

static inline int ceil_div(int a, int b) { return (a + b - 1) / b; }

extern "C" void kernel_launch(void* const* d_in, const int* in_sizes, int n_in,
                              void* d_out, int out_size, void* d_ws, size_t ws_size,
                              hipStream_t stream) {
  (void)out_size; (void)ws_size;
  Ptrs p;
  auto F = [&](int i) { return (const float*)d_in[i]; };
  if (n_in > 0 && in_sizes[0] == NBATCH * SEQL * ENC) {
    // insertion order: x, revin_w, revin_b, lin1..lin4 (w,b), m1..m4
    p.x = F(0); p.revin_w = F(1); p.revin_b = F(2);
    p.lin1_w = F(3);  p.lin1_b = F(4);  p.lin2_w = F(5);  p.lin2_b = F(6);
    p.lin3_w = F(7);  p.lin3_b = F(8);  p.lin4_w = F(9);  p.lin4_b = F(10);
    int i = 11;
    MP* ms[4] = { &p.m1, &p.m2, &p.m3, &p.m4 };
    for (int k = 0; k < 4; ++k) {
      MP* m = ms[k];
      m->in_w = F(i++); m->conv_w = F(i++); m->conv_b = F(i++); m->xproj_w = F(i++);
      m->dt_w = F(i++); m->dt_b = F(i++); m->A_log = F(i++); m->D = F(i++); m->out_w = F(i++);
    }
  } else {
    // jax tree (sorted-key) order
    p.lin1_b = F(0); p.lin1_w = F(1); p.lin2_b = F(2); p.lin2_w = F(3);
    p.lin3_b = F(4); p.lin3_w = F(5); p.lin4_b = F(6); p.lin4_w = F(7);
    int i = 8;
    MP* ms[4] = { &p.m1, &p.m2, &p.m3, &p.m4 };
    for (int k = 0; k < 4; ++k) {
      MP* m = ms[k];
      m->A_log = F(i++); m->D = F(i++); m->conv_b = F(i++); m->conv_w = F(i++);
      m->dt_b = F(i++); m->dt_w = F(i++); m->in_w = F(i++); m->out_w = F(i++); m->xproj_w = F(i++);
    }
    p.revin_b = F(i++); p.revin_w = F(i++); p.x = F(i++);
  }

  // --- workspace arena ---
  char* base = (char*)d_ws; size_t off = 0;
  auto alloc = [&](size_t bytes) -> void* {
    void* r = base + off; off = (off + bytes + 255) & ~(size_t)255; return r;
  };
  const int R = RROWS;
  _Float16* w_lin1 = (_Float16*)alloc(sizeof(_Float16) * 256 * 512);
  _Float16* w_lin2 = (_Float16*)alloc(sizeof(_Float16) * 128 * 256);
  _Float16* w_lin3 = (_Float16*)alloc(sizeof(_Float16) * 256 * 128);
  _Float16* w_lin4 = (_Float16*)alloc(sizeof(_Float16) * 96  * 512);
  _Float16* w3_in  = (_Float16*)alloc(sizeof(_Float16) * 512 * 256);
  _Float16* w3_xp  = (_Float16*)alloc(sizeof(_Float16) * NDBC * 256); // padded from 272 rows
  _Float16* w3_out = (_Float16*)alloc(sizeof(_Float16) * 256 * 256);
  _Float16* w2_in  = (_Float16*)alloc(sizeof(_Float16) * 256 * 128);
  _Float16* w2_xp  = (_Float16*)alloc(sizeof(_Float16) * NDBC * 128); // padded from 264 rows
  _Float16* w2_out = (_Float16*)alloc(sizeof(_Float16) * 128 * 128);
  float*    meanv  = (float*)alloc(sizeof(float) * R);
  float*    stdv   = (float*)alloc(sizeof(float) * R);
  _Float16* xn16   = (_Float16*)alloc(sizeof(_Float16) * R * 512);
  float*    h1     = (float*)alloc(sizeof(float) * R * 256);
  _Float16* h1h    = (_Float16*)alloc(sizeof(_Float16) * R * 256);
  float*    xz     = (float*)alloc(sizeof(float) * R * 512);   // m3 then m2
  float*    xa     = (float*)alloc(sizeof(float) * R * 256);
  _Float16* xah    = (_Float16*)alloc(sizeof(_Float16) * R * 256);
  float*    dbc    = (float*)alloc(sizeof(float) * R * NDBC);
  _Float16* yh     = (_Float16*)alloc(sizeof(_Float16) * R * 256);
  float*    x3     = (float*)alloc(sizeof(float) * R * 256);
  float*    h2     = (float*)alloc(sizeof(float) * R * 128);
  _Float16* h2h    = (_Float16*)alloc(sizeof(_Float16) * R * 128);
  float*    x1     = (float*)alloc(sizeof(float) * R * 128);
  float*    x2     = (float*)alloc(sizeof(float) * R * 128);
  _Float16* s16    = (_Float16*)alloc(sizeof(_Float16) * R * 128);
  float*    t3     = (float*)alloc(sizeof(float) * R * 256);
  _Float16* cat16  = (_Float16*)alloc(sizeof(_Float16) * R * 512);
  float*    outlin = (float*)alloc(sizeof(float) * R * PREDL);

  auto cvt = [&](_Float16* dst, const float* src, int ndst, int nsrc) {
    k_cvt_f16<<<ceil_div(ndst, 256), 256, 0, stream>>>(dst, src, ndst, nsrc);
  };
  cvt(w_lin1, p.lin1_w, 256*512, 256*512);
  cvt(w_lin2, p.lin2_w, 128*256, 128*256);
  cvt(w_lin3, p.lin3_w, 256*128, 256*128);
  cvt(w_lin4, p.lin4_w,  96*512,  96*512);
  cvt(w3_in,  p.m3.in_w,   512*256, 512*256);
  cvt(w3_xp,  p.m3.xproj_w, NDBC*256, 272*256);  // zero-pad rows 272..287
  cvt(w3_out, p.m3.out_w,  256*256, 256*256);
  cvt(w2_in,  p.m2.in_w,   256*128, 256*128);
  cvt(w2_xp,  p.m2.xproj_w, NDBC*128, 264*128);  // zero-pad rows 264..287
  cvt(w2_out, p.m2.out_w,  128*128, 128*128);

  auto gemm = [&](const _Float16* A, const _Float16* W, const float* bias,
                  float* Cf, _Float16* Ch, int M, int N, int K) {
    if (N % 64 == 0) {
      int tiles = ceil_div(M, 32) * (N / 64);
      k_gemm_wmma<2, 4><<<tiles, 32, 0, stream>>>(A, W, bias, Cf, Ch, M, N, K);
    } else {           // N multiple of 32 (288, 96)
      int tiles = ceil_div(M, 32) * (N / 32);
      k_gemm_wmma<2, 2><<<tiles, 32, 0, stream>>>(A, W, bias, Cf, Ch, M, N, K);
    }
  };

  // 1) RevIN + transpose -> xn16 (R x 512)
  k_revin<<<R, 32, 0, stream>>>(p.x, p.revin_w, p.revin_b, xn16, meanv, stdv);
  // 2) h1 = xn @ lin1^T + b         (R x 256)
  gemm(xn16, w_lin1, p.lin1_b, h1, h1h, R, 256, 512);
  // 3) m3 (d=256, L=1)
  gemm(h1h, w3_in, nullptr, xz, nullptr, R, 512, 256);
  k_conv_silu<<<ceil_div(R*256, 256), 256, 0, stream>>>(xz, p.m3.conv_w, p.m3.conv_b, xa, xah, R, 256);
  gemm(xah, w3_xp, nullptr, dbc, nullptr, R, NDBC, 256);
  k_finalize_l1<<<R, 32, 0, stream>>>(dbc, xa, xz, p.m3.dt_w, p.m3.dt_b, p.m3.D, yh, R, 256, 16);
  gemm(yh, w3_out, nullptr, x3, nullptr, R, 256, 256);
  // 4) m4 (d=1, L=256) + x3 -> concat right half (f16)
  k_mamba_seq<<<R, 32, 0, stream>>>(h1, 256, p.m4.in_w, p.m4.conv_w, p.m4.conv_b,
                                    p.m4.xproj_w, p.m4.dt_w, p.m4.dt_b, p.m4.A_log,
                                    p.m4.D, p.m4.out_w, x3, nullptr, cat16, 512, 256, R);
  // 5) h2 = h1 @ lin2^T + b         (R x 128)
  gemm(h1h, w_lin2, p.lin2_b, h2, h2h, R, 128, 256);
  // 6) m1 (d=1, L=128) -> x1
  k_mamba_seq<<<R, 32, 0, stream>>>(h2, 128, p.m1.in_w, p.m1.conv_w, p.m1.conv_b,
                                    p.m1.xproj_w, p.m1.dt_w, p.m1.dt_b, p.m1.A_log,
                                    p.m1.D, p.m1.out_w, nullptr, x1, nullptr, 0, 0, R);
  // 7) m2 (d=128, L=1)
  gemm(h2h, w2_in, nullptr, xz, nullptr, R, 256, 128);
  k_conv_silu<<<ceil_div(R*128, 256), 256, 0, stream>>>(xz, p.m2.conv_w, p.m2.conv_b, xa, xah, R, 128);
  gemm(xah, w2_xp, nullptr, dbc, nullptr, R, NDBC, 128);
  k_finalize_l1<<<R, 32, 0, stream>>>(dbc, xa, xz, p.m2.dt_w, p.m2.dt_b, p.m2.D, yh, R, 128, 8);
  gemm(yh, w2_out, nullptr, x2, nullptr, R, 128, 128);
  // 8) s = x1 + h2 + x2 ; t = s @ lin3^T + b ; concat left = t + h1
  k_sum3_f16<<<ceil_div(R*128, 256), 256, 0, stream>>>(x1, h2, x2, s16, R*128);
  gemm(s16, w_lin3, p.lin3_b, t3, nullptr, R, 256, 128);
  k_addres<<<ceil_div(R*256, 256), 256, 0, stream>>>(t3, h1, cat16, R);
  // 9) out = concat @ lin4^T + b    (R x 96)
  gemm(cat16, w_lin4, p.lin4_b, outlin, nullptr, R, 96, 512);
  // 10) de-RevIN + transpose to (B, 96, C)
  k_final<<<ceil_div(NBATCH*PREDL*ENC, 256), 256, 0, stream>>>(
      outlin, p.revin_w, p.revin_b, meanv, stdv, (float*)d_out);
}
